// CrossAttention_63221918597757
// MI455X (gfx1250) — compile-verified, matrix operand order checked by reading, not run on hardware
//
#include <hip/hip_runtime.h>
#include <hip/hip_bf16.h>

// ---- problem constants -----------------------------------------------------
#define BATCH   16
#define NQ      4096
#define JCTX    77
#define JPAD    96      // 6 n-tiles of 16 / 3 k-steps of 32
#define HEADS   8
#define DHEAD   40
#define DPAD    64      // 2 k-steps of 32
#define DOUT    48      // 3 n-tiles of 16 for attn@V
#define QDIM    320
#define CDIM    768
#define INNER   320

typedef _Float16 v16h __attribute__((ext_vector_type(16)));
typedef _Float16 v8h  __attribute__((ext_vector_type(8)));
typedef float    v8f  __attribute__((ext_vector_type(8)));
typedef float    f4   __attribute__((ext_vector_type(4)));

// D = A*B + C,  16x16 tile, K=32, f16 inputs, f32 accumulate
__device__ __forceinline__ v8f wmma16(v16h a, v16h b, v8f c) {
  return __builtin_amdgcn_wmma_f32_16x16x32_f16(
      /*neg_a=*/false, a, /*neg_b=*/false, b,
      /*c_mod=*/(short)0, c, /*reuse_a=*/false, /*reuse_b=*/false);
}

// A/B fragment from an f16 row (K contiguous).  lane layout (wave32):
// halves [kbase+h8, +8) in v[0..7], [kbase+16+h8, +8) in v[8..15], h8=(lane>>4)*8
__device__ __forceinline__ v16h frag_f16(const _Float16* row, int kbase, int h8) {
  v8h lo = *(const v8h*)(row + kbase + h8);
  v8h hi = *(const v8h*)(row + kbase + 16 + h8);
  v16h r;
#pragma unroll
  for (int i = 0; i < 8; ++i) { r[i] = lo[i]; r[i + 8] = hi[i]; }
  return r;
}

// same fragment but converting f32 memory -> f16 on the fly
__device__ __forceinline__ v16h frag_f32(const float* row, int kbase, int h8) {
  f4 a0 = *(const f4*)(row + kbase + h8);
  f4 a1 = *(const f4*)(row + kbase + h8 + 4);
  f4 b0 = *(const f4*)(row + kbase + 16 + h8);
  f4 b1 = *(const f4*)(row + kbase + 16 + h8 + 4);
  v16h r;
#pragma unroll
  for (int i = 0; i < 4; ++i) {
    r[i]      = (_Float16)a0[i];
    r[i + 4]  = (_Float16)a1[i];
    r[i + 8]  = (_Float16)b0[i];
    r[i + 12] = (_Float16)b1[i];
  }
  return r;
}

// ---- small utility kernels -------------------------------------------------
__global__ void k_cvt(const float* __restrict__ src, _Float16* __restrict__ dst, int n) {
  int i = blockIdx.x * blockDim.x + threadIdx.x;
  if (i < n) dst[i] = (_Float16)src[i];
}

__global__ void k_fill0(v8h* __restrict__ dst, long n8) {
  long i = (long)blockIdx.x * blockDim.x + threadIdx.x;
  long stride = (long)gridDim.x * blockDim.x;
  v8h z = {};
  for (; i < n8; i += stride) dst[i] = z;
}

// ---- Q projection: Qp[row][h][DPAD] = (x @ Wq^T) split heads, f16 ---------
__global__ void k_proj_q(const float* __restrict__ x, const _Float16* __restrict__ W,
                         _Float16* __restrict__ Qp) {
  int job = blockIdx.x * 8 + (threadIdx.x >> 5);          // 20480 wave jobs
  int lane = threadIdx.x & 31, ln = lane & 15, hl = lane >> 4, h8 = hl * 8;
  int mt = job / 5, nt = job % 5;                          // 4096 m-tiles x 5 n64-tiles
  const float* arow = x + (size_t)(mt * 16 + ln) * QDIM;
  v8f acc[4] = {};
#pragma unroll
  for (int ks = 0; ks < 10; ++ks) {                        // K = 320
    v16h a = frag_f32(arow, 32 * ks, h8);
#pragma unroll
    for (int t = 0; t < 4; ++t) {
      v16h b = frag_f16(W + (size_t)(nt * 64 + t * 16 + ln) * QDIM, 32 * ks, h8);
      acc[t] = wmma16(a, b, acc[t]);
    }
  }
#pragma unroll
  for (int t = 0; t < 4; ++t)
#pragma unroll
    for (int r = 0; r < 8; ++r) {
      int row = mt * 16 + r + 8 * hl;
      int c = nt * 64 + t * 16 + ln, h = c / DHEAD, d = c % DHEAD;
      Qp[((size_t)row * HEADS + h) * DPAD + d] = (_Float16)acc[t][r];
    }
}

// ---- K/V projection. mode 0: K16[b][h][j][DPAD]; mode 1: Vt16[b][h][d][JPAD]
__global__ void k_proj_kv(const float* __restrict__ ctx, const _Float16* __restrict__ W,
                          _Float16* __restrict__ dst, int mode) {
  int job = blockIdx.x * 8 + (threadIdx.x >> 5);
  if (job >= 77 * 5) return;                               // 1232 rows = 77 m-tiles
  int lane = threadIdx.x & 31, ln = lane & 15, hl = lane >> 4, h8 = hl * 8;
  int mt = job / 5, nt = job % 5;
  const float* arow = ctx + (size_t)(mt * 16 + ln) * CDIM;
  v8f acc[4] = {};
#pragma unroll
  for (int ks = 0; ks < 24; ++ks) {                        // K = 768
    v16h a = frag_f32(arow, 32 * ks, h8);
#pragma unroll
    for (int t = 0; t < 4; ++t) {
      v16h b = frag_f16(W + (size_t)(nt * 64 + t * 16 + ln) * CDIM, 32 * ks, h8);
      acc[t] = wmma16(a, b, acc[t]);
    }
  }
#pragma unroll
  for (int t = 0; t < 4; ++t)
#pragma unroll
    for (int r = 0; r < 8; ++r) {
      int row = mt * 16 + r + 8 * hl;                      // < 1232 exactly
      int bb = row / JCTX, j = row % JCTX;
      int c = nt * 64 + t * 16 + ln, h = c / DHEAD, d = c % DHEAD;
      _Float16 v = (_Float16)acc[t][r];
      if (mode == 0)
        dst[(((size_t)bb * HEADS + h) * JPAD + j) * DPAD + d] = v;
      else
        dst[(((size_t)bb * HEADS + h) * DOUT + d) * JPAD + j] = v;
    }
}

// ---- fused attention: one wave = one (b,h,16-row q tile) -------------------
__global__ void k_attn(const _Float16* __restrict__ Qp, const _Float16* __restrict__ K16,
                       const _Float16* __restrict__ Vt, const unsigned char* __restrict__ mask,
                       _Float16* __restrict__ O16) {
  __shared__ _Float16 sattn[8][16][JPAD];                  // 24 KB, one 16x96 tile per wave
  int wid = threadIdx.x >> 5, lane = threadIdx.x & 31;
  int ln = lane & 15, hl = lane >> 4, h8 = hl * 8;
  int job = blockIdx.x * 8 + wid;                          // 32768 jobs
  int bh = job >> 8, nt = job & 255;                       // 256 q-tiles per (b,h)
  int b = bh >> 3, h = bh & 7;

  // Q fragments (K = DPAD = 64 -> 2 k-steps)
  const _Float16* qrow = Qp + ((size_t)(b * NQ + nt * 16 + ln) * HEADS + h) * DPAD;
  v16h qa0 = frag_f16(qrow, 0, h8);
  v16h qa1 = frag_f16(qrow, 32, h8);

  // sim = Q K^T over 6 j-tiles
  const _Float16* kbase = K16 + (size_t)(b * HEADS + h) * JPAD * DPAD;
  v8f s[6] = {};
#pragma unroll
  for (int jt = 0; jt < 6; ++jt) {
    const _Float16* krow = kbase + (size_t)(jt * 16 + ln) * DPAD;
    s[jt] = wmma16(qa0, frag_f16(krow, 0, h8), s[jt]);
    s[jt] = wmma16(qa1, frag_f16(krow, 32, h8), s[jt]);
  }

  // scale + mask (lane owns column j = jt*16+ln for every row)
  const float scale = 0.15811388300841897f;                // 40^-0.5
  bool ok[6];
#pragma unroll
  for (int jt = 0; jt < 6; ++jt) {
    int j = jt * 16 + ln;
    ok[jt] = (j < JCTX) && (mask[b * JCTX + j] != 0);
  }
#pragma unroll
  for (int jt = 0; jt < 6; ++jt)
#pragma unroll
    for (int r = 0; r < 8; ++r)
      s[jt][r] = ok[jt] ? s[jt][r] * scale : -3.0e38f;

  // softmax over j: reduce across the 16 lanes of each half-wave
  float inv[8];
#pragma unroll
  for (int r = 0; r < 8; ++r) {
    float m = -3.0e38f;
#pragma unroll
    for (int jt = 0; jt < 6; ++jt) m = fmaxf(m, s[jt][r]);
#pragma unroll
    for (int off = 1; off < 16; off <<= 1) m = fmaxf(m, __shfl_xor(m, off, 32));
    float sum = 0.f;
#pragma unroll
    for (int jt = 0; jt < 6; ++jt) {
      float e = __expf(s[jt][r] - m);
      s[jt][r] = e;
      sum += e;
    }
#pragma unroll
    for (int off = 1; off < 16; off <<= 1) sum += __shfl_xor(sum, off, 32);
    inv[r] = 1.0f / sum;
  }

  // C-layout -> A-layout via LDS (f16 probabilities)
#pragma unroll
  for (int jt = 0; jt < 6; ++jt)
#pragma unroll
    for (int r = 0; r < 8; ++r)
      sattn[wid][r + 8 * hl][jt * 16 + ln] = (_Float16)(s[jt][r] * inv[r]);
  __syncthreads();

  v16h pa[3];
#pragma unroll
  for (int ks = 0; ks < 3; ++ks) pa[ks] = frag_f16(&sattn[wid][ln][0], 32 * ks, h8);

  // out = attn @ V  (B operand = V^T rows, K = j)
  const _Float16* vbase = Vt + (size_t)(b * HEADS + h) * DOUT * JPAD;
#pragma unroll
  for (int dt = 0; dt < 3; ++dt) {
    v8f o = {};
    const _Float16* vrow = vbase + (size_t)(dt * 16 + ln) * JPAD;
#pragma unroll
    for (int ks = 0; ks < 3; ++ks) o = wmma16(pa[ks], frag_f16(vrow, 32 * ks, h8), o);
    int d = dt * 16 + ln;
    if (d < DHEAD) {
#pragma unroll
      for (int r = 0; r < 8; ++r) {
        int row = b * NQ + nt * 16 + r + 8 * hl;
        O16[(size_t)row * INNER + h * DHEAD + d] = (_Float16)o[r];
      }
    }
  }
}

// ---- output projection: out = O16 @ Wo^T + bo (f32) ------------------------
__global__ void k_out(const _Float16* __restrict__ O, const _Float16* __restrict__ W,
                      const float* __restrict__ bo, float* __restrict__ out) {
  int job = blockIdx.x * 8 + (threadIdx.x >> 5);
  int lane = threadIdx.x & 31, ln = lane & 15, hl = lane >> 4, h8 = hl * 8;
  int mt = job / 5, nt = job % 5;
  const _Float16* arow = O + (size_t)(mt * 16 + ln) * INNER;
  v8f acc[4] = {};
#pragma unroll
  for (int ks = 0; ks < 10; ++ks) {
    v16h a = frag_f16(arow, 32 * ks, h8);
#pragma unroll
    for (int t = 0; t < 4; ++t) {
      v16h b = frag_f16(W + (size_t)(nt * 64 + t * 16 + ln) * INNER, 32 * ks, h8);
      acc[t] = wmma16(a, b, acc[t]);
    }
  }
#pragma unroll
  for (int t = 0; t < 4; ++t)
#pragma unroll
    for (int r = 0; r < 8; ++r) {
      int row = mt * 16 + r + 8 * hl;
      int c = nt * 64 + t * 16 + ln;
      out[(size_t)row * QDIM + c] = acc[t][r] + bo[c];
    }
}

// ---- host side -------------------------------------------------------------
extern "C" void kernel_launch(void* const* d_in, const int* in_sizes, int n_in,
                              void* d_out, int out_size, void* d_ws, size_t ws_size,
                              hipStream_t stream) {
  const float* x    = (const float*)d_in[0];
  const float* ctx  = (const float*)d_in[1];
  const unsigned char* mask = (const unsigned char*)d_in[2];
  const float* Wq   = (const float*)d_in[3];
  const float* Wk   = (const float*)d_in[4];
  const float* Wv   = (const float*)d_in[5];
  const float* Wo   = (const float*)d_in[6];
  const float* bo   = (const float*)d_in[7];

  // workspace layout (f16)
  _Float16* Qp   = (_Float16*)d_ws;                                  // B*NQ*H*DPAD
  _Float16* K16  = Qp  + (size_t)BATCH * NQ * HEADS * DPAD;          // B*H*JPAD*DPAD
  _Float16* Vt16 = K16 + (size_t)BATCH * HEADS * JPAD * DPAD;        // B*H*DOUT*JPAD
  _Float16* O16  = Vt16 + (size_t)BATCH * HEADS * DOUT * JPAD;       // B*NQ*INNER
  _Float16* Wq16 = O16 + (size_t)BATCH * NQ * INNER;
  _Float16* Wk16 = Wq16 + (size_t)INNER * QDIM;
  _Float16* Wv16 = Wk16 + (size_t)INNER * CDIM;
  _Float16* Wo16 = Wv16 + (size_t)INNER * CDIM;

  // weight conversion
  k_cvt<<<(INNER * QDIM + 255) / 256, 256, 0, stream>>>(Wq, Wq16, INNER * QDIM);
  k_cvt<<<(INNER * CDIM + 255) / 256, 256, 0, stream>>>(Wk, Wk16, INNER * CDIM);
  k_cvt<<<(INNER * CDIM + 255) / 256, 256, 0, stream>>>(Wv, Wv16, INNER * CDIM);
  k_cvt<<<(INNER * QDIM + 255) / 256, 256, 0, stream>>>(Wo, Wo16, INNER * QDIM);

  // zero the padded Q/K/Vt region (contiguous)
  long fillN = (long)BATCH * NQ * HEADS * DPAD
             + (long)BATCH * HEADS * JPAD * DPAD
             + (long)BATCH * HEADS * DOUT * JPAD;
  k_fill0<<<4096, 256, 0, stream>>>((v8h*)Qp, fillN / 8);

  // projections
  k_proj_q<<<(4096 * 5) / 8, 256, 0, stream>>>(x, Wq16, Qp);
  k_proj_kv<<<(77 * 5 + 7) / 8, 256, 0, stream>>>(ctx, Wk16, K16, 0);
  k_proj_kv<<<(77 * 5 + 7) / 8, 256, 0, stream>>>(ctx, Wv16, Vt16, 1);

  // fused attention
  k_attn<<<(BATCH * HEADS * (NQ / 16)) / 8, 256, 0, stream>>>(Qp, K16, Vt16, mask, O16);

  // output projection + bias
  k_out<<<(4096 * 5) / 8, 256, 0, stream>>>(O16, Wo16, bo, (float*)d_out);
}
